// ProteinGraphConvModule_15032385536148
// MI455X (gfx1250) — matrix operand order repaired and checked by reading, not compiled, not run on hardware
//
#include <hip/hip_runtime.h>
#include <math.h>
#include <stdint.h>

// ---------------- CDNA5 types ----------------
typedef __bf16 bf16x16 __attribute__((ext_vector_type(16)));
typedef __bf16 bf16x2  __attribute__((ext_vector_type(2)));
typedef float  f32x8   __attribute__((ext_vector_type(8)));
typedef unsigned int u32x4 __attribute__((ext_vector_type(4)));
typedef int          i32x4 __attribute__((ext_vector_type(4)));
typedef int          i32x8 __attribute__((ext_vector_type(8)));

#if __has_builtin(__builtin_amdgcn_tensor_load_to_lds)
#define USE_TDM 1
#else
#define USE_TDM 0
#endif

__device__ __forceinline__ unsigned f2bf1(float a) {
  unsigned ua = __float_as_uint(a);
  ua = (ua + 0x7FFFu + ((ua >> 16) & 1u)) >> 16;   // RNE
  return ua;
}
__device__ __forceinline__ unsigned f2bf2(float a, float b) {
#if __has_builtin(__builtin_amdgcn_cvt_pk_bf16_f32)
  union { bf16x2 h; unsigned u; } cv;
  cv.h = __builtin_amdgcn_cvt_pk_bf16_f32(a, b);   // single v_cvt_pk_bf16_f32
  return cv.u;
#else
  return f2bf1(a) | (f2bf1(b) << 16);
#endif
}

__device__ __forceinline__ void atomAddF(float* p, float v) {
  __hip_atomic_fetch_add(p, v, __ATOMIC_RELAXED, __HIP_MEMORY_SCOPE_AGENT);
}

__device__ __forceinline__ void waitTensorCnt0() {
#if __has_builtin(__builtin_amdgcn_s_wait_tensorcnt)
  __builtin_amdgcn_s_wait_tensorcnt(0);
#else
  asm volatile("s_wait_tensorcnt 0" ::: "memory");
#endif
}

// Escape the LDS pointer + memory clobber: the TDM writes LDS behind the
// compiler's back, so loads from this buffer must not be folded or hoisted.
__device__ __forceinline__ void ldsPublish(void* p) {
  asm volatile("" : : "r"(p) : "memory");
}

// ---------------- utility kernels ----------------
__global__ void fill_zero_kernel(float* p, int n) {
  int i = blockIdx.x * blockDim.x + threadIdx.x;
  if (i < n) p[i] = 0.0f;
}

// transpose-convert: src f32 [K][Nn] row-major  ->  dst bf16 [Nn][K]
__global__ void wconv_kernel(const float* __restrict__ src, unsigned short* __restrict__ dst,
                             int K, int Nn) {
  int i = blockIdx.x * blockDim.x + threadIdx.x;
  if (i >= K * Nn) return;
  int n = i / K, k = i - n * K;
  dst[i] = (unsigned short)f2bf1(src[k * Nn + n]);
}

__global__ void relu_inplace_kernel(float* p, int n) {
  int i = blockIdx.x * blockDim.x + threadIdx.x;
  if (i < n) p[i] = fmaxf(p[i], 0.0f);
}

// ---------------- WMMA GEMM: C = act(A_f32[M,Kd] x W + bias) ----------------
// W supplied pre-transposed+converted: Wt bf16 [Nn][Kd] (row n = column n of W)
// block 256 threads (8 waves); WG tile 128x64; wave tile 32x32 (2x2 WMMA tiles)
// B tile fetched by the Tensor Data Mover (one descriptor per K-step) when available.
#define TM 128
#define TN 64
#define TK 32

__global__ __launch_bounds__(256)
void gemm_bf16_wmma_kernel(const float* __restrict__ A, const unsigned short* __restrict__ Wt,
                           const float* __restrict__ bias, float* __restrict__ C,
                           int M, int Kd, int Nn, int act) {
  // B first: assume it lands at LDS offset 0 for the TDM descriptor.
  __shared__ __align__(16) unsigned BlsU[TN * TK / 2];  // 64 rows x 16 dwords (bf16 pairs)
  __shared__ __align__(16) unsigned AlsU[TM * TK / 2];  // 128 rows x 16 dwords

  const int tid  = threadIdx.x;
  const int lane = tid & 31, wave = tid >> 5;
  const int wm = wave & 3, wn = wave >> 2;       // 4x2 wave grid
  const int m0 = blockIdx.x * TM, n0 = blockIdx.y * TN;
  const int half = lane >> 4, l15 = lane & 15;

  f32x8 acc[2][2];
  for (int rt = 0; rt < 2; ++rt)
    for (int ct = 0; ct < 2; ++ct)
      for (int j = 0; j < 8; ++j) acc[rt][ct][j] = 0.0f;

  for (int k0 = 0; k0 < Kd; k0 += TK) {
    __syncthreads();
#if USE_TDM
    // --- B tile via Tensor Data Mover: 64x32 bf16 slab of Wt into LDS offset 0 ---
    if (wave == 0) {
      unsigned long long ga = (unsigned long long)(Wt + (size_t)n0 * Kd + k0);
      u32x4 g0 = { 1u,                       // count=1, is_restore=0
                   0u,                       // lds_addr (BlsU at LDS offset 0)
                   (unsigned)ga,
                   (unsigned)((ga >> 32) & 0x1FFFFFFu) | (2u << 30) };  // addr hi | type=2
      i32x8 g1;
      g1[0] = 0x10000;                                        // data_size = 1 (2 bytes)
      g1[1] = (int)(((unsigned)Kd & 0xFFFFu) << 16);          // tensor_dim0 lo16
      g1[2] = (int)(((unsigned)Kd >> 16) & 0xFFFFu) |
              (int)(((unsigned)Nn & 0xFFFFu) << 16);          // dim0 hi16 | tensor_dim1 lo16
      g1[3] = (int)(((unsigned)Nn >> 16) & 0xFFFFu) |
              (int)(32u << 16);                               // dim1 hi16 | tile_dim0 = 32
      g1[4] = 64;                                             // tile_dim1 = 64, tile_dim2 = 0
      g1[5] = Kd;                                             // tensor_dim0_stride lo32
      g1[6] = 0;
      g1[7] = 0;
      i32x4 z4 = { 0, 0, 0, 0 };
#if defined(__clang_major__) && (__clang_major__ >= 23)
      i32x8 z8 = { 0, 0, 0, 0, 0, 0, 0, 0 };
      __builtin_amdgcn_tensor_load_to_lds(g0, g1, z4, z4, z8, 0);
#else
      __builtin_amdgcn_tensor_load_to_lds(g0, g1, z4, z4, 0);
#endif
      waitTensorCnt0();
    }
#endif
    // --- stage A tile (f32 -> bf16) : 128x32 ---
    {
      const int col  = (tid & 7) * 4;          // element col
      const int colD = (tid & 7) * 2;          // dword col in LDS
      for (int r = 0; r < 4; ++r) {
        int row = (tid >> 3) + r * 32;
        int gm  = m0 + row;
        uint2 pk;
        if (gm < M) {
          const float* ap = A + (size_t)gm * Kd + k0 + col;
          if (k0 + TK < Kd) __builtin_prefetch(ap + TK, 0, 1);  // global_prefetch next K-slab
          const float4 f = *(const float4*)ap;
          pk.x = f2bf2(f.x, f.y);
          pk.y = f2bf2(f.z, f.w);
        } else {
          pk.x = 0u; pk.y = 0u;
        }
        *(uint2*)&AlsU[row * 16 + colD] = pk;
      }
    }
#if !USE_TDM
    // --- stage B tile (already bf16, N-major) : 64x32 ---
    {
      int n  = tid >> 2;
      int kc = (tid & 3) * 8;
      const uint4 w = *(const uint4*)(Wt + (size_t)(n0 + n) * Kd + k0 + kc);
      *(uint4*)&BlsU[n * 16 + (tid & 3) * 4] = w;
    }
#endif
    __syncthreads();
    ldsPublish((void*)BlsU);   // TDM wrote this buffer: escape + clobber => reload fragments

    // --- build fragments per documented 16-bit WMMA layouts ---
    union FB { uint4 q[2]; bf16x16 v; };
    FB afr[2], bfr[2];
    for (int rt = 0; rt < 2; ++rt) {
      const uint4* p = (const uint4*)&AlsU[(wm * 32 + rt * 16 + l15) * 16];
      afr[rt].q[0] = p[half];        // K = half*8 .. +7
      afr[rt].q[1] = p[2 + half];    // K = 16+half*8 .. +7
    }
    for (int ct = 0; ct < 2; ++ct) {
      const uint4* p = (const uint4*)&BlsU[(wn * 32 + ct * 16 + l15) * 16];
      bfr[ct].q[0] = p[half];        // K = half*8 .. +7
      bfr[ct].q[1] = p[2 + half];    // K = 16+half*8 .. +7
    }
    for (int rt = 0; rt < 2; ++rt)
      for (int ct = 0; ct < 2; ++ct)
        acc[rt][ct] = __builtin_amdgcn_wmma_f32_16x16x32_bf16(
            false, afr[rt].v, false, bfr[ct].v, (short)0, acc[rt][ct], false, false);
  }

  // --- epilogue: bias + optional ReLU, per C layout (VGPR j -> M = half*8 + j) ---
  for (int ct = 0; ct < 2; ++ct) {
    int n = n0 + wn * 32 + ct * 16 + l15;
    float bv = bias ? bias[n] : 0.0f;
    for (int rt = 0; rt < 2; ++rt) {
      int mb = m0 + wm * 32 + rt * 16 + half * 8;
      for (int j = 0; j < 8; ++j) {
        int m = mb + j;
        if (m < M) {
          float v = acc[rt][ct][j] + bv;
          if (act) v = fmaxf(v, 0.0f);
          C[(size_t)m * Nn + n] = v;
        }
      }
    }
  }
}

// ---------------- edge attention (one wave per edge) ----------------
// scores[h,g] = <q[tgt,h,:], k[src,g,:]>/sqrt(D) + bias[h]; softmax over g;
// msg[h,:] = sum_g attn[h,g] * v[src,g,:]; scatter-add to out[tgt].
__global__ __launch_bounds__(256)
void edge_attn_kernel(const float* __restrict__ Q, const float* __restrict__ Kf,
                      const float* __restrict__ Vf, const int* __restrict__ src,
                      const int* __restrict__ tgt, const float* __restrict__ bias,
                      float* __restrict__ out, int E, int F, int D, float scale) {
  __shared__ float attn_s[8][64];
  const int wave = threadIdx.x >> 5, lane = threadIdx.x & 31;
  const int e = blockIdx.x * 8 + wave;
  if (e >= E) return;

  const int s = src[e], t = tgt[e];
  const float* q = Q  + (size_t)t * F;
  const float* k = Kf + (size_t)s * F;
  const float* v = Vf + (size_t)s * F;

  // two (h,g) pairs per lane
  float sc[2];
  for (int i = 0; i < 2; ++i) {
    int p = lane + i * 32;
    int h = p >> 3, g = p & 7;
    const float* qr = q + h * D;
    const float* kr = k + g * D;
    float a = 0.0f;
    for (int d = 0; d < D; ++d) a = fmaf(qr[d], kr[d], a);
    sc[i] = a * scale + bias[h];
  }
  // softmax over g: butterfly across 8-lane groups
  for (int i = 0; i < 2; ++i) {
    float m = sc[i];
    for (int off = 1; off < 8; off <<= 1) m = fmaxf(m, __shfl_xor(m, off, 32));
    float ex = __expf(sc[i] - m);
    float su = ex;
    for (int off = 1; off < 8; off <<= 1) su += __shfl_xor(su, off, 32);
    attn_s[wave][lane + i * 32] = ex / su;
  }
  asm volatile("s_wait_dscnt 0" ::: "memory");  // LDS stores visible to this wave's loads

  // msg + scatter
  for (int idx = lane; idx < F; idx += 32) {
    int h = idx / D, d = idx - h * D;
    const float* aw = &attn_s[wave][h * 8];
    float m = 0.0f;
    for (int g = 0; g < 8; ++g) m = fmaf(aw[g], v[g * D + d], m);
    atomAddF(&out[(size_t)t * F + idx], m);
  }
}

// ---------------- graph pooling / GRU / output tail ----------------
__global__ void pool_accum_kernel(const float* __restrict__ h, const int* __restrict__ batch,
                                  float* __restrict__ pool, float* __restrict__ counts, int N) {
  int i = blockIdx.x * blockDim.x + threadIdx.x;
  if (i >= N * 256) return;
  int n = i >> 8, c = i & 255;
  int b = batch[n];
  atomAddF(&pool[b * 256 + c], h[(size_t)n * 256 + c]);
  if (c == 0) atomAddF(&counts[b], 1.0f);
}

__global__ void superf_lin_kernel(const float* __restrict__ pool, const float* __restrict__ counts,
                                  const float* __restrict__ W, const float* __restrict__ bias,
                                  float* __restrict__ s1, int B) {
  int i = blockIdx.x * blockDim.x + threadIdx.x;
  if (i >= B * 256) return;
  int b = i >> 8, c = i & 255;
  float inv = 1.0f / counts[b];
  float a = bias[c];
  for (int k = 0; k < 256; ++k) a = fmaf(pool[b * 256 + k] * inv, W[k * 256 + c], a);
  s1[i] = fmaxf(a, 0.0f);
}

__global__ void gru_kernel(const float* __restrict__ x, const float* __restrict__ Wih,
                           const float* __restrict__ bih, const float* __restrict__ bhh,
                           float* __restrict__ sf, int B) {
  int i = blockIdx.x * blockDim.x + threadIdx.x;
  if (i >= B * 256) return;
  int b = i >> 8, c = i & 255;
  const float* xr = x + b * 256;
  float g0 = bih[c], g1 = bih[256 + c], g2 = bih[512 + c];
  for (int k = 0; k < 256; ++k) {
    float xv = xr[k];
    g0 = fmaf(xv, Wih[(size_t)c * 256 + k], g0);
    g1 = fmaf(xv, Wih[(size_t)(256 + c) * 256 + k], g1);
    g2 = fmaf(xv, Wih[(size_t)(512 + c) * 256 + k], g2);
  }
  float r = 1.0f / (1.0f + __expf(-(g0 + bhh[c])));
  float z = 1.0f / (1.0f + __expf(-(g1 + bhh[256 + c])));
  float nn = tanhf(g2 + r * bhh[512 + c]);
  sf[i] = (1.0f - z) * nn;  // h0 == 0
}

__global__ void final_kernel(const float* __restrict__ h, const float* __restrict__ sf,
                             const int* __restrict__ batch, float* __restrict__ out, int N) {
  int i = blockIdx.x * blockDim.x + threadIdx.x;
  if (i >= N * 512) return;
  int n = i >> 9, c = i & 511;
  float val;
  if (c < 256) {
    int i2 = c & ~1;
    float freq = __expf(-9.210340371976184f * (float)i2 * (1.0f / 256.0f));
    float ang  = (float)n * freq;
    float pe   = (c & 1) ? __cosf(ang) : __sinf(ang);
    val = h[(size_t)n * 256 + c] + pe;
  } else {
    val = sf[batch[n] * 256 + (c - 256)];
  }
  out[i] = val;
}

// ---------------- host orchestration ----------------
extern "C" void kernel_launch(void* const* d_in, const int* in_sizes, int n_in,
                              void* d_out, int out_size, void* d_ws, size_t ws_size,
                              hipStream_t stream) {
  (void)n_in; (void)out_size; (void)ws_size;
  const int N = in_sizes[0] / 64;   // 6000
  const int E = in_sizes[1] / 2;    // 64000
  const int B = 32;
  const int finA[3]  = {512, 512, 512};
  const int foutA[3] = {512, 512, 256};

  const float* x        = (const float*)d_in[0];
  const int*   ei       = (const int*)d_in[1];
  const int*   batch    = (const int*)d_in[2];
  const float* embW     = (const float*)d_in[4];
  const float* embB     = (const float*)d_in[5];
  const float* sftW     = (const float*)d_in[33];
  const float* sftB     = (const float*)d_in[34];
  const float* gWih     = (const float*)d_in[35];
  const float* gbih     = (const float*)d_in[37];
  const float* gbhh     = (const float*)d_in[38];
  const int* src = ei;          // row 0 = source
  const int* tgt = ei + E;      // row 1 = target

  uint8_t* ws = (uint8_t*)d_ws;
  size_t off = 0;
  auto carve = [&](size_t bytes) -> void* {
    void* p = ws + off;
    off = (off + bytes + 255) & ~(size_t)255;
    return p;
  };

  unsigned short* wtEmb = (unsigned short*)carve((size_t)512 * 64 * 2);
  unsigned short* wtl[3], *wtq[3], *wtk[3], *wtv[3];
  for (int l = 0; l < 3; ++l) {
    wtl[l] = (unsigned short*)carve((size_t)foutA[l] * finA[l] * 2);
    wtq[l] = (unsigned short*)carve((size_t)foutA[l] * foutA[l] * 2);
    wtk[l] = (unsigned short*)carve((size_t)foutA[l] * foutA[l] * 2);
    wtv[l] = (unsigned short*)carve((size_t)foutA[l] * foutA[l] * 2);
  }
  float* bufH = (float*)carve((size_t)N * 512 * 4);
  float* bufX = (float*)carve((size_t)N * 512 * 4);
  float* bufQ = (float*)carve((size_t)N * 512 * 4);
  float* bufK = (float*)carve((size_t)N * 512 * 4);
  float* bufV = (float*)carve((size_t)N * 512 * 4);
  float* pool   = (float*)carve((size_t)B * 256 * 4);
  float* counts = (float*)carve((size_t)B * 4);
  float* s1     = (float*)carve((size_t)B * 256 * 4);
  float* sf     = (float*)carve((size_t)B * 256 * 4);

  auto launch_wconv = [&](const float* srcW, unsigned short* dst, int K, int Nn) {
    int tot = K * Nn;
    hipLaunchKernelGGL(wconv_kernel, dim3((tot + 255) / 256), dim3(256), 0, stream, srcW, dst, K, Nn);
  };
  auto launch_gemm = [&](const float* A, const unsigned short* Wt, const float* bias,
                         float* C, int M, int Kd, int Nn, int act) {
    hipLaunchKernelGGL(gemm_bf16_wmma_kernel, dim3((M + TM - 1) / TM, Nn / TN), dim3(256), 0,
                       stream, A, Wt, bias, C, M, Kd, Nn, act);
  };
  auto launch_fill = [&](float* p, int n) {
    hipLaunchKernelGGL(fill_zero_kernel, dim3((n + 255) / 256), dim3(256), 0, stream, p, n);
  };

  // 1) weight conversion (one-time per call; tiny)
  launch_wconv(embW, wtEmb, 64, 512);
  for (int l = 0; l < 3; ++l) {
    const float* Wl = (const float*)d_in[6 + 9 * l + 0];
    const float* Wq = (const float*)d_in[6 + 9 * l + 2];
    const float* Wk = (const float*)d_in[6 + 9 * l + 4];
    const float* Wv = (const float*)d_in[6 + 9 * l + 6];
    launch_wconv(Wl, wtl[l], finA[l], foutA[l]);
    launch_wconv(Wq, wtq[l], foutA[l], foutA[l]);
    launch_wconv(Wk, wtk[l], foutA[l], foutA[l]);
    launch_wconv(Wv, wtv[l], foutA[l], foutA[l]);
  }

  // 2) embedding: h = relu(x @ embW + b)
  launch_gemm(x, wtEmb, embB, bufH, N, 64, 512, 1);

  // 3) three conv layers
  for (int l = 0; l < 3; ++l) {
    const int fin = finA[l], fout = foutA[l], D = fout / 8;
    const float* bl = (const float*)d_in[6 + 9 * l + 1];
    const float* bq = (const float*)d_in[6 + 9 * l + 3];
    const float* bk = (const float*)d_in[6 + 9 * l + 5];
    const float* bv = (const float*)d_in[6 + 9 * l + 7];
    const float* ab = (const float*)d_in[6 + 9 * l + 8];

    launch_gemm(bufH, wtl[l], bl, bufX, N, fin, fout, 0);     // x = h @ Wl + bl
    launch_gemm(bufX, wtq[l], bq, bufQ, N, fout, fout, 0);    // node-level Q
    launch_gemm(bufX, wtk[l], bk, bufK, N, fout, fout, 0);    // node-level K
    launch_gemm(bufX, wtv[l], bv, bufV, N, fout, fout, 0);    // node-level V

    launch_fill(bufH, N * fout);                              // agg buffer (reuse h)
    float scale = rsqrtf((float)D);
    hipLaunchKernelGGL(edge_attn_kernel, dim3((E + 7) / 8), dim3(256), 0, stream,
                       bufQ, bufK, bufV, src, tgt, ab, bufH, E, fout, D, scale);
    hipLaunchKernelGGL(relu_inplace_kernel, dim3((N * fout + 255) / 256), dim3(256), 0, stream,
                       bufH, N * fout);                       // h = relu(agg)
  }

  // 4) per-graph mean pool + super-feature linear + GRU
  launch_fill(pool, B * 256);
  launch_fill(counts, B);
  hipLaunchKernelGGL(pool_accum_kernel, dim3((N * 256 + 255) / 256), dim3(256), 0, stream,
                     bufH, batch, pool, counts, N);
  hipLaunchKernelGGL(superf_lin_kernel, dim3((B * 256 + 255) / 256), dim3(256), 0, stream,
                     pool, counts, sftW, sftB, s1, B);
  hipLaunchKernelGGL(gru_kernel, dim3((B * 256 + 255) / 256), dim3(256), 0, stream,
                     s1, gWih, gbih, gbhh, sf, B);

  // 5) positional encoding + concat
  hipLaunchKernelGGL(final_kernel, dim3((N * 512 + 255) / 256), dim3(256), 0, stream,
                     bufH, sf, batch, (float*)d_out, N);
}
